// GatGraphClassifier_74371653698083
// MI455X (gfx1250) — compile-verified, matrix operand order checked by reading, not run on hardware
//
#include <hip/hip_runtime.h>
#include <hip/hip_bf16.h>

#define N_NODES   10000
#define N_EDGES   160000
#define N_FEATS   9
#define D_DIM     128
#define H_HEADS   4
#define C_DIM     128
#define HC        512        // H_HEADS * C_DIM
#define N_LAYERS  3
#define N_CLASSES 10
#define NPAD      10016      // 626*16, multiple of 32
#define MTILES    626        // NPAD/16
#define NTILES    32         // HC/16
#define NBLK      4          // N-tiles per wave (16x64 output slab)
#define KTILES    4          // D_DIM/32
#define WPACK_ELEMS (N_LAYERS*KTILES*NTILES*512)  // 196608

typedef __attribute__((ext_vector_type(16))) __bf16 v16bf;
typedef __attribute__((ext_vector_type(8)))  float  v8f;

union Frag16 { uint4 u[2]; v16bf v; };

__device__ __forceinline__ unsigned short f32_to_bf16(float f) {
    unsigned int u = __float_as_uint(f);
    u += 0x7FFFu + ((u >> 16) & 1u);   // round-to-nearest-even
    return (unsigned short)(u >> 16);
}

__device__ __forceinline__ void atomicMaxFloat(float* addr, float value) {
    if (value >= 0.0f)
        atomicMax((int*)addr, __float_as_int(value));
    else
        atomicMin((unsigned int*)addr, __float_as_uint(value));
}

// ---- Pack W [L,128,512] f32 -> WMMA B-fragments, bf16, lane-contiguous ----
// Fragment (l,kt,nt): 32 lanes x 16 bf16. lane holds column n = nt*16 + (lane&15),
// K(j,lane) = kt*32 + (lane>=16?8:0) + (j<8 ? j : j+8)  (16-bit B 32x16 ISA layout).
__global__ void k_pack_w(const float* __restrict__ W, unsigned short* __restrict__ wpack) {
    int t = blockIdx.x * blockDim.x + threadIdx.x;
    if (t >= WPACK_ELEMS) return;
    int j    =  t        & 15;
    int lane = (t >> 4)  & 31;
    int nt   = (t >> 9)  & 31;
    int kt   = (t >> 14) & 3;
    int l    =  t >> 16;
    int base = (lane >= 16) ? 8 : 0;
    int K = kt * 32 + base + ((j < 8) ? j : j + 8);
    int n = nt * 16 + (lane & 15);
    wpack[t] = f32_to_bf16(W[((size_t)l * D_DIM + K) * HC + n]);
}

// ---- AtomEncoder: x[n,c] = sum_f emb[f, feat[n,f], c] ----
__global__ void k_atom_encoder(const int* __restrict__ feats, const float* __restrict__ emb,
                               float* __restrict__ xf, unsigned short* __restrict__ xbf) {
    int t = blockIdx.x * blockDim.x + threadIdx.x;
    if (t >= N_NODES * D_DIM) return;
    int n = t >> 7, c = t & 127;
    float s = 0.0f;
#pragma unroll
    for (int f = 0; f < N_FEATS; ++f) {
        int v = feats[n * N_FEATS + f];
        s += emb[((size_t)f * 128 + v) * D_DIM + c];
    }
    xf[t]  = s;
    xbf[t] = f32_to_bf16(s);
}

// ---- Zero padding rows of x (rows N_NODES..NPAD-1) ----
__global__ void k_zero_pad(float* __restrict__ xf, unsigned short* __restrict__ xbf) {
    int t = blockIdx.x * blockDim.x + threadIdx.x;
    int npad = (NPAD - N_NODES) * D_DIM;
    if (t >= npad) return;
    xf [N_NODES * D_DIM + t] = 0.0f;
    xbf[N_NODES * D_DIM + t] = 0;
}

// ---- Per-layer init: acc=0, segmax=-inf, segsum=0 ----
__global__ void k_layer_init(float* __restrict__ acc, float* __restrict__ segmax,
                             float* __restrict__ segsum) {
    int t = blockIdx.x * blockDim.x + threadIdx.x;
    if (t < NPAD * D_DIM) acc[t] = 0.0f;
    if (t < NPAD * H_HEADS) {
        segmax[t] = -__builtin_huge_valf();
        segsum[t] = 0.0f;
    }
}

// ---- WMMA GEMM: h[NPAD,512] = xbf[NPAD,128] @ Wpack(layer).
// One wave per 16x64 output slab: A fragment loaded once per k-step and fed to
// 4 back-to-back WMMAs (register-blocked over N). 16 v_wmma per wave. ----
__global__ __launch_bounds__(32) void k_gemm_wmma(const unsigned short* __restrict__ xbf,
                                                  const unsigned short* __restrict__ wpack,
                                                  float* __restrict__ hbuf, int layer) {
    int mtile = blockIdx.x;
    int ntg   = blockIdx.y;                // group of NBLK n-tiles
    int lane  = threadIdx.x;
    int row   = mtile * 16 + (lane & 15);
    int base  = (lane >> 4) << 3;          // 0 or 8
    v8f acc[NBLK] = {};
#pragma unroll
    for (int kt = 0; kt < KTILES; ++kt) {
        // A fragment: row-major bf16 activations; K = kt*32 + base + {0..7, 16..23}
        Frag16 a;
        const uint4* pa = (const uint4*)(xbf + (size_t)row * D_DIM + kt * 32 + base);
        a.u[0] = pa[0];     // K = base+0 .. base+7
        a.u[1] = pa[2];     // K = base+16 .. base+23
#pragma unroll
        for (int nn = 0; nn < NBLK; ++nn) {
            int ntile = ntg * NBLK + nn;
            Frag16 b;
            const uint4* pb = (const uint4*)(wpack +
                ((((size_t)layer * KTILES + kt) * NTILES + ntile) << 9) + lane * 16);
            b.u[0] = pb[0];
            b.u[1] = pb[1];
            acc[nn] = __builtin_amdgcn_wmma_f32_16x16x32_bf16(false, a.v, false, b.v,
                                                              (short)0, acc[nn], false, false);
        }
    }
    // D layout: VGPR r -> M = r + (lane>=16 ? 8 : 0), N = lane&15
    int mbase = mtile * 16 + ((lane >> 4) << 3);
#pragma unroll
    for (int nn = 0; nn < NBLK; ++nn) {
        int col = (ntg * NBLK + nn) * 16 + (lane & 15);
#pragma unroll
        for (int r = 0; r < 8; ++r)
            hbuf[(size_t)(mbase + r) * HC + col] = acc[nn][r];
    }
}

// ---- alpha_src/alpha_dst: per (node, head) dot of length 128, wave-reduced ----
__global__ __launch_bounds__(128) void k_alpha(const float* __restrict__ hbuf,
                                               const float* __restrict__ a_s,
                                               const float* __restrict__ a_d,
                                               float* __restrict__ asrc,
                                               float* __restrict__ adst) {
    int n    = blockIdx.x;                 // node
    int head = threadIdx.x >> 5;
    int lane = threadIdx.x & 31;
    float ss = 0.0f, sd = 0.0f;
#pragma unroll
    for (int i = 0; i < 4; ++i) {
        int c = lane + i * 32;
        float hv = hbuf[(size_t)n * HC + head * C_DIM + c];
        ss += hv * a_s[head * C_DIM + c];
        sd += hv * a_d[head * C_DIM + c];
    }
#pragma unroll
    for (int o = 16; o > 0; o >>= 1) {
        ss += __shfl_xor(ss, o);
        sd += __shfl_xor(sd, o);
    }
    if (lane == 0) {
        asrc[n * H_HEADS + head] = ss;
        adst[n * H_HEADS + head] = sd;
    }
}

// ---- Edge logits: e = leaky(asrc[src]+adst[dst]); segment max over dst ----
__global__ void k_edge_logits(const int* __restrict__ edges, const float* __restrict__ asrc,
                              const float* __restrict__ adst, float* __restrict__ ebuf,
                              float* __restrict__ segmax) {
    int e = blockIdx.x * blockDim.x + threadIdx.x;
    if (e >= N_EDGES) return;
    int src = edges[e];
    int dst = edges[N_EDGES + e];
#pragma unroll
    for (int h = 0; h < H_HEADS; ++h) {
        float v = asrc[src * H_HEADS + h] + adst[dst * H_HEADS + h];
        v = (v >= 0.0f) ? v : 0.2f * v;     // LeakyReLU(0.2)
        ebuf[e * H_HEADS + h] = v;
        atomicMaxFloat(&segmax[dst * H_HEADS + h], v);
    }
}

// ---- Edge exp + segment sum ----
__global__ void k_edge_exp(const int* __restrict__ edges, float* __restrict__ ebuf,
                           const float* __restrict__ segmax, float* __restrict__ segsum) {
    int e = blockIdx.x * blockDim.x + threadIdx.x;
    if (e >= N_EDGES) return;
    int dst = edges[N_EDGES + e];
#pragma unroll
    for (int h = 0; h < H_HEADS; ++h) {
        float m  = segmax[dst * H_HEADS + h];
        float ex = __expf(ebuf[e * H_HEADS + h] - m);
        ebuf[e * H_HEADS + h] = ex;
        atomicAdd(&segsum[dst * H_HEADS + h], ex);
    }
}

// ---- Edge aggregation (head-mean folded): acc[dst,c] += 0.25*sum_h alpha*h[src,h,c] ----
__global__ void k_edge_aggregate(const int* __restrict__ edges, const float* __restrict__ ebuf,
                                 const float* __restrict__ segsum, const float* __restrict__ hbuf,
                                 float* __restrict__ acc) {
    long long t = (long long)blockIdx.x * blockDim.x + threadIdx.x;
    if (t >= (long long)N_EDGES * C_DIM) return;
    int e = (int)(t >> 7);
    int c = (int)(t & 127);
    int src = edges[e];
    int dst = edges[N_EDGES + e];
    float s = 0.0f;
#pragma unroll
    for (int h = 0; h < H_HEADS; ++h) {
        float alpha = ebuf[e * H_HEADS + h] / (segsum[dst * H_HEADS + h] + 1e-16f);
        s += alpha * hbuf[(size_t)src * HC + h * C_DIM + c];
    }
    atomicAdd(&acc[(size_t)dst * C_DIM + c], 0.25f * s);
}

// ---- Finalize layer: + bias, optional GELU(tanh), emit f32 + bf16 ----
__global__ void k_finalize(const float* __restrict__ acc, const float* __restrict__ bias_l,
                           float* __restrict__ xf, unsigned short* __restrict__ xbf,
                           int apply_gelu) {
    int t = blockIdx.x * blockDim.x + threadIdx.x;
    if (t >= N_NODES * D_DIM) return;
    int c = t & 127;
    float v = acc[t] + bias_l[c];
    if (apply_gelu) {
        float u = 0.7978845608028654f * (v + 0.044715f * v * v * v);
        v = 0.5f * v * (1.0f + tanhf(u));
    }
    xf[t]  = v;
    xbf[t] = f32_to_bf16(v);
}

// ---- Pooling: pooled[c] += x[n,c] ----
__global__ void k_zero128(float* __restrict__ p) {
    if (threadIdx.x < 128) p[threadIdx.x] = 0.0f;
}
__global__ void k_pool(const float* __restrict__ xf, float* __restrict__ pooled) {
    int t = blockIdx.x * blockDim.x + threadIdx.x;
    if (t >= N_NODES * D_DIM) return;
    atomicAdd(&pooled[t & 127], xf[t]);
}

// ---- Classifier: out = (pooled/N) @ Wc + bc ----
__global__ void k_classify(const float* __restrict__ pooled, const float* __restrict__ Wc,
                           const float* __restrict__ bc, float* __restrict__ out) {
    int t = threadIdx.x;
    if (t >= N_CLASSES) return;
    float s = bc[t];
    const float invN = 1.0f / (float)N_NODES;
#pragma unroll 4
    for (int d = 0; d < D_DIM; ++d)
        s += pooled[d] * invN * Wc[d * N_CLASSES + t];
    out[t] = s;
}

extern "C" void kernel_launch(void* const* d_in, const int* in_sizes, int n_in,
                              void* d_out, int out_size, void* d_ws, size_t ws_size,
                              hipStream_t stream) {
    const int*   feats   = (const int*)d_in[0];
    const int*   edges   = (const int*)d_in[1];
    const float* emb     = (const float*)d_in[2];
    const float* W       = (const float*)d_in[3];
    const float* att_src = (const float*)d_in[4];
    const float* att_dst = (const float*)d_in[5];
    const float* bias    = (const float*)d_in[6];
    const float* Wc      = (const float*)d_in[7];
    const float* bc      = (const float*)d_in[8];
    float* out = (float*)d_out;

    char* ws = (char*)d_ws;
    size_t off = 0;
    auto alloc = [&](size_t bytes) -> void* {
        void* p = ws + off;
        off = (off + bytes + 255) & ~(size_t)255;
        return p;
    };
    unsigned short* wpack  = (unsigned short*)alloc((size_t)WPACK_ELEMS * 2);
    unsigned short* xbf    = (unsigned short*)alloc((size_t)NPAD * D_DIM * 2);
    float* xf     = (float*)alloc((size_t)NPAD * D_DIM * 4);
    float* hbuf   = (float*)alloc((size_t)NPAD * HC * 4);
    float* asrc   = (float*)alloc((size_t)NPAD * H_HEADS * 4);
    float* adst   = (float*)alloc((size_t)NPAD * H_HEADS * 4);
    float* ebuf   = (float*)alloc((size_t)N_EDGES * H_HEADS * 4);
    float* segmax = (float*)alloc((size_t)NPAD * H_HEADS * 4);
    float* segsum = (float*)alloc((size_t)NPAD * H_HEADS * 4);
    float* accb   = (float*)alloc((size_t)NPAD * D_DIM * 4);
    float* pooled = (float*)alloc(128 * 4);

    const int TB = 256;

    k_pack_w<<<(WPACK_ELEMS + TB - 1) / TB, TB, 0, stream>>>(W, wpack);
    k_atom_encoder<<<(N_NODES * D_DIM + TB - 1) / TB, TB, 0, stream>>>(feats, emb, xf, xbf);
    k_zero_pad<<<((NPAD - N_NODES) * D_DIM + TB - 1) / TB, TB, 0, stream>>>(xf, xbf);

    for (int l = 0; l < N_LAYERS; ++l) {
        k_layer_init<<<(NPAD * D_DIM + TB - 1) / TB, TB, 0, stream>>>(accb, segmax, segsum);
        dim3 gemm_grid(MTILES, NTILES / NBLK);
        k_gemm_wmma<<<gemm_grid, 32, 0, stream>>>(xbf, wpack, hbuf, l);
        k_alpha<<<N_NODES, 128, 0, stream>>>(hbuf,
                                             att_src + (size_t)l * H_HEADS * C_DIM,
                                             att_dst + (size_t)l * H_HEADS * C_DIM,
                                             asrc, adst);
        k_edge_logits<<<(N_EDGES + TB - 1) / TB, TB, 0, stream>>>(edges, asrc, adst, ebuf, segmax);
        k_edge_exp<<<(N_EDGES + TB - 1) / TB, TB, 0, stream>>>(edges, ebuf, segmax, segsum);
        long long agg = (long long)N_EDGES * C_DIM;
        k_edge_aggregate<<<(unsigned)((agg + TB - 1) / TB), TB, 0, stream>>>(edges, ebuf, segsum,
                                                                             hbuf, accb);
        k_finalize<<<(N_NODES * D_DIM + TB - 1) / TB, TB, 0, stream>>>(
            accb, bias + (size_t)l * C_DIM, xf, xbf, (l < N_LAYERS - 1) ? 1 : 0);
    }

    k_zero128<<<1, 128, 0, stream>>>(pooled);
    k_pool<<<(N_NODES * D_DIM + TB - 1) / TB, TB, 0, stream>>>(xf, pooled);
    k_classify<<<1, 32, 0, stream>>>(pooled, Wc, bc, out);
}
